// AdaptivePADReHAR_584115553028
// MI455X (gfx1250) — compile-verified, matrix operand order checked by reading, not run on hardware
//
#include <hip/hip_runtime.h>
#include <hip/hip_bf16.h>
#include <math.h>

// ---------------------------------------------------------------------------
// Model constants (match reference)
// ---------------------------------------------------------------------------
constexpr int Ln    = 4;
constexpr int Hn    = 256;
constexpr int Dn    = 3;
constexpr int Kk    = 11;
constexpr int CINn  = 9;
constexpr int NCLSn = 18;
constexpr int Bsz   = 64;
constexpr int Tn    = 1024;
constexpr float TEMP = 5.0f;
constexpr float EPSc = 1e-5f;

// All activations use channels-last [B][T][C] layout so every WMMA operand is
// K-contiguous in memory (no transposes anywhere in the GEMM path).

typedef __attribute__((ext_vector_type(16))) _Float16 v16h;
typedef __attribute__((ext_vector_type(8)))  _Float16 v8h;
typedef __attribute__((ext_vector_type(8)))  float    v8f;
typedef __attribute__((ext_vector_type(4)))  unsigned int v4u;
typedef __attribute__((ext_vector_type(4)))  int      v4i;
typedef __attribute__((ext_vector_type(8)))  int      v8i;

__device__ __forceinline__ float gelu_f(float x) {
  return 0.5f * x * (1.0f + erff(x * 0.7071067811865476f));
}

__device__ __forceinline__ float block_reduce_sum(float v, float* sbuf) {
  const int tid = threadIdx.x;
  sbuf[tid] = v;
  __syncthreads();
  for (int s = 128; s > 0; s >>= 1) {
    if (tid < s) sbuf[tid] += sbuf[tid + s];
    __syncthreads();
  }
  float r = sbuf[0];
  __syncthreads();
  return r;
}

// ---------------------------------------------------------------------------
// TDM: DMA a 2D f16 tile (rows x 32 halves, row stride = kdim halves) from
// global into LDS, padding each 64B row with 16B so the LDS layout matches the
// 80B-stride fragment layout. Descriptor per cdna5 ISA 08_async_tensor §8:
//   g0: count=1 | lds_addr | global_addr(57b) | type=2
//   g1: data_size=2B, pad_enable, pad_interval=3 (16 DW), pad_amount=3 (4 DW),
//       tensor_dim0=kdim, tensor_dim1=trows, tile_dim0=32, tile_dim1=rows,
//       tensor_dim0_stride=kdim
//   g2/g3: zero (2-D tensor)
// ---------------------------------------------------------------------------
#define GEMM_KT 32
#define LDS_STR 40  // halves (80B stride): 16B-aligned groups, banks x20 mod 64
#define LDS_TILE_B (128 * LDS_STR * 2)  // bytes per tile buffer (10240)

__device__ __forceinline__ void tdm_load_tile(
    unsigned lds_off, const _Float16* gptr, int kdim, int rows, int trows) {
  unsigned long long ga = (unsigned long long)(size_t)gptr;
  v4u g0;
  g0.x = 1u;                                          // count=1, user mode
  g0.y = lds_off;                                     // LDS byte address
  g0.z = (unsigned)(ga & 0xFFFFFFFFull);              // global_addr[31:0]
  g0.w = (unsigned)((ga >> 32) & 0x01FFFFFFull) | (2u << 30);  // [56:32]|type=2
  v8i g1;
  g1[0] = (1 << 16)            // data_size = 2 bytes
        | (1 << 20)            // pad_enable
        | (3 << 22)            // pad_interval: 16 DWORDs (one 64B row)
        | (3 << 25);           // pad_amount: 4 DWORDs (16B)
  g1[1] = (kdim & 0xFFFF) << 16;          // tensor_dim0[15:0]
  g1[2] = (trows & 0xFFFF) << 16;         // tensor_dim0[31:16]=0, tensor_dim1 lo
  g1[3] = (GEMM_KT << 16);                // tensor_dim1 hi=0, tile_dim0 = 32
  g1[4] = rows & 0xFFFF;                  // tile_dim1 = rows, tile_dim2 = 0
  g1[5] = kdim;                           // tensor_dim0_stride[31:0]
  g1[6] = 0;                              // stride hi, dim1_stride lo (unused 2D)
  g1[7] = 0;
  v4i z4 = {0, 0, 0, 0};
#if defined(__clang_major__) && (__clang_major__ >= 23)
  v8i z8 = {0, 0, 0, 0, 0, 0, 0, 0};
  __builtin_amdgcn_tensor_load_to_lds(g0, g1, z4, z4, z8, 0);
#else
  __builtin_amdgcn_tensor_load_to_lds(g0, g1, z4, z4, 0);
#endif
}

// ---------------------------------------------------------------------------
// fp32 -> f16 weight conversion (once per launch; stays L2-resident)
// ---------------------------------------------------------------------------
__global__ __launch_bounds__(256) void k_cvt(
    const float* __restrict__ src, _Float16* __restrict__ dst, int n) {
  int i = blockIdx.x * 256 + threadIdx.x;
  if (i < n) dst[i] = (_Float16)src[i];
}

// ---------------------------------------------------------------------------
// Stem: X[b,t,h] = sum_c x[b,c,t] * in_w[h,c] + in_b[h]  (+ f16 mirror)
// grid (T, B), block 256 (= h); writes coalesced along h.
// ---------------------------------------------------------------------------
__global__ __launch_bounds__(256) void k_pw_in(
    const float* __restrict__ x, const float* __restrict__ in_w,
    const float* __restrict__ in_b, float* __restrict__ X,
    _Float16* __restrict__ Xh) {
  const int h = threadIdx.x;
  const int t = blockIdx.x;
  const int b = blockIdx.y;
  float acc = in_b[h];
#pragma unroll
  for (int c = 0; c < CINn; ++c)
    acc += x[((size_t)b * CINn + c) * Tn + t] * in_w[h * CINn + c];
  size_t idx = ((size_t)b * Tn + t) * Hn + h;
  X[idx] = acc;
  Xh[idx] = (_Float16)acc;
}

// ---------------------------------------------------------------------------
// Gate features, stage 1: per (b, t-chunk) partial sums over t for each c.
// grid (16, B), block 256 (= c). FEATP[b][chunk][3][256].
// ---------------------------------------------------------------------------
#define FT_CH 64
__global__ __launch_bounds__(256) void k_feat1(
    const float* __restrict__ X, float* __restrict__ FEATP) {
  const int c = threadIdx.x;
  const int ch = blockIdx.x;
  const int b = blockIdx.y;
  const size_t bb = (size_t)b * Tn;
  const int t0 = ch * FT_CH;
  float s0 = 0.f, s1 = 0.f, s2 = 0.f;
  for (int t = t0; t < t0 + FT_CH; ++t) {
    float v = X[(bb + t) * Hn + c];
    s0 += v;
    s1 += fabsf(v);
    if (t > 0) s2 += fabsf(v - X[(bb + t - 1) * Hn + c]);
  }
  size_t o = ((size_t)b * 16 + ch) * 3 * Hn + c;
  FEATP[o] = s0;
  FEATP[o + Hn] = s1;
  FEATP[o + 2 * Hn] = s2;
}

// Stage 2: combine 16 chunks -> FEAT[b][3*256]. grid (B), block 256.
__global__ __launch_bounds__(256) void k_feat2(
    const float* __restrict__ FEATP, float* __restrict__ FEAT) {
  const int c = threadIdx.x;
  const int b = blockIdx.x;
  float s0 = 0.f, s1 = 0.f, s2 = 0.f;
  for (int ch = 0; ch < 16; ++ch) {
    size_t o = ((size_t)b * 16 + ch) * 3 * Hn + c;
    s0 += FEATP[o];
    s1 += FEATP[o + Hn];
    s2 += FEATP[o + 2 * Hn];
  }
  FEAT[(size_t)b * 3 * Hn + c]          = s0 / (float)Tn;
  FEAT[(size_t)b * 3 * Hn + Hn + c]     = s1 / (float)Tn;
  FEAT[(size_t)b * 3 * Hn + 2 * Hn + c] = s2 / (float)(Tn - 1);
}

// ---------------------------------------------------------------------------
// Gate logits + softmax(logits/TEMP). grid 1, block 64 (thread = batch)
// ---------------------------------------------------------------------------
__global__ __launch_bounds__(64) void k_logits(
    const float* __restrict__ FEAT, const float* __restrict__ gw,
    const float* __restrict__ gb, float* __restrict__ GATE) {
  const int b = threadIdx.x;
  if (b >= Bsz) return;
  const float* f = FEAT + (size_t)b * 3 * Hn;
  float l[Dn];
#pragma unroll
  for (int d = 0; d < Dn; ++d) {
    float s = gb[d];
    const float* wrow = gw + (size_t)d * 3 * Hn;
    for (int j = 0; j < 3 * Hn; ++j) s += f[j] * wrow[j];
    l[d] = s / TEMP;
  }
  float mx = fmaxf(l[0], fmaxf(l[1], l[2]));
  float e0 = expf(l[0] - mx), e1 = expf(l[1] - mx), e2 = expf(l[2] - mx);
  float inv = 1.0f / (e0 + e1 + e2);
  GATE[b * Dn + 0] = e0 * inv;
  GATE[b * Dn + 1] = e1 * inv;
  GATE[b * Dn + 2] = e2 * inv;
}

// ---------------------------------------------------------------------------
// WMMA GEMM (channels-last): OUT[b,t,m] = sum_k IN[b,t,k] * W[m,k] + bias[m]
// A-matrix = activation tile (rows = t, K contiguous in global)
// B-matrix = weight rows     (cols = m, K contiguous in global)
// Tile staging via the Tensor Data Mover with double-buffered LDS: wave 0
// DMAs activation tiles, wave 1 weight tiles. Stage n+1's DMA is issued
// before computing stage n; s_wait_tensorcnt 1 (TDM completes in order per
// wave) overlaps DMA with WMMA. TDM row padding produces the 80B LDS stride.
// Fragments are two 16B ds_load_b128 per operand (K groups {8h..8h+7},
// {16+8h..16+8h+7}, h = lane>=16). 8 waves, each 64(t) x 32(m) = 4x2 WMMA
// 16x16x32 f16 tiles, f32 accumulate.
// act: 1 = exact GELU.  outMode: 0 = fp32 OUTf, 1 = f16 OUTh.
// ---------------------------------------------------------------------------
__global__ __launch_bounds__(256) void k_pw_gemm(
    const _Float16* __restrict__ INh, const _Float16* __restrict__ Wh,
    const float* __restrict__ bias, float* __restrict__ OUTf,
    _Float16* __restrict__ OUTh, int M, int Kdim, int act, int outMode) {
  __shared__ __align__(16) _Float16 Ats[2][128][LDS_STR];
  __shared__ __align__(16) _Float16 Wts[2][128][LDS_STR];

  const int b   = blockIdx.z;
  const int t0  = blockIdx.x * 128;
  const int m0  = blockIdx.y * 128;
  const int tid = threadIdx.x;
  const int lane = tid & 31;
  const int wave = tid >> 5;
  const int wv   = __builtin_amdgcn_readfirstlane(wave);  // scalar wave id
  const int wt   = (wave >> 2) * 64;  // wave t-offset: 0 or 64
  const int wm   = (wave & 3) * 32;   // wave m-offset: 0,32,64,96
  const int lrow = lane & 15;
  const int hsel = lane >> 4;

  const _Float16* INb = INh + (size_t)b * Tn * Kdim;
  const unsigned ldsA0 = (unsigned)(size_t)&Ats[0][0][0];
  const unsigned ldsA1 = (unsigned)(size_t)&Ats[1][0][0];
  const unsigned ldsW0 = (unsigned)(size_t)&Wts[0][0][0];
  const unsigned ldsW1 = (unsigned)(size_t)&Wts[1][0][0];

  // Per-thread prefetch coordinates (warm L2 two stages ahead of the TDM).
  const int r0 = tid >> 2, q0 = tid & 3;

  v8f acc[4][2];
#pragma unroll
  for (int ti = 0; ti < 4; ++ti)
#pragma unroll
    for (int tj = 0; tj < 2; ++tj)
#pragma unroll
      for (int r = 0; r < 8; ++r) acc[ti][tj][r] = 0.0f;

  // Prologue: DMA the first k-stage into buffer 0.
  if (wv == 0)      tdm_load_tile(ldsA0, INb + (size_t)t0 * Kdim, Kdim, 128, Tn);
  else if (wv == 1) tdm_load_tile(ldsW0, Wh + (size_t)m0 * Kdim, Kdim, 128, M);

  int p = 0;
  for (int k0 = 0; k0 < Kdim; k0 += GEMM_KT) {
    if (k0 + GEMM_KT < Kdim) {
      // Issue next stage into the other buffer, then wait for current stage
      // only (<=1 outstanding): overlaps the next DMA with this compute.
      if (wv == 0) {
        tdm_load_tile(p ? ldsA0 : ldsA1,
                      INb + (size_t)t0 * Kdim + (k0 + GEMM_KT), Kdim, 128, Tn);
      } else if (wv == 1) {
        tdm_load_tile(p ? ldsW0 : ldsW1,
                      Wh + (size_t)m0 * Kdim + (k0 + GEMM_KT), Kdim, 128, M);
      } else if (wv == 2 && k0 + 2 * GEMM_KT < Kdim) {
        __builtin_prefetch(INb + (size_t)(t0 + r0) * Kdim + (k0 + 2 * GEMM_KT + q0 * 8), 0, 1);
        __builtin_prefetch(Wh + (size_t)(m0 + r0) * Kdim + (k0 + 2 * GEMM_KT + q0 * 8), 0, 1);
      }
      if (wv < 2) __builtin_amdgcn_s_wait_tensorcnt(1);
    } else {
      if (wv < 2) __builtin_amdgcn_s_wait_tensorcnt(0);
    }
    __syncthreads();

    v16h af[4], bf[2];
#pragma unroll
    for (int ti = 0; ti < 4; ++ti) {
      int row = wt + ti * 16 + lrow;
      v8h lo = *(const v8h*)&Ats[p][row][hsel * 8];
      v8h hi = *(const v8h*)&Ats[p][row][16 + hsel * 8];
      af[ti] = __builtin_shufflevector(lo, hi, 0, 1, 2, 3, 4, 5, 6, 7,
                                       8, 9, 10, 11, 12, 13, 14, 15);
    }
#pragma unroll
    for (int tj = 0; tj < 2; ++tj) {
      int col = wm + tj * 16 + lrow;
      v8h lo = *(const v8h*)&Wts[p][col][hsel * 8];
      v8h hi = *(const v8h*)&Wts[p][col][16 + hsel * 8];
      bf[tj] = __builtin_shufflevector(lo, hi, 0, 1, 2, 3, 4, 5, 6, 7,
                                       8, 9, 10, 11, 12, 13, 14, 15);
    }
#pragma unroll
    for (int ti = 0; ti < 4; ++ti)
#pragma unroll
      for (int tj = 0; tj < 2; ++tj)
        acc[ti][tj] = __builtin_amdgcn_wmma_f32_16x16x32_f16(
            false, af[ti], false, bf[tj], (short)0, acc[ti][tj], false, false);
    __syncthreads();  // all waves done with buffer p before it is re-DMA'd
    p ^= 1;
  }

  // Epilogue. D layout: lane -> column (m), VGPR r -> row t + hsel*8 + r.
#pragma unroll
  for (int tj = 0; tj < 2; ++tj) {
    int m = m0 + wm + tj * 16 + lrow;
    float bm = bias[m];
#pragma unroll
    for (int ti = 0; ti < 4; ++ti) {
      int tbase = t0 + wt + ti * 16 + hsel * 8;
#pragma unroll
      for (int r = 0; r < 8; ++r) {
        int t = tbase + r;
        float v = acc[ti][tj][r] + bm;
        if (act) v = gelu_f(v);
        size_t oidx = ((size_t)b * Tn + t) * M + m;
        if (outMode == 0) OUTf[oidx] = v;
        else              OUTh[oidx] = (_Float16)v;
      }
    }
  }
}

// ---------------------------------------------------------------------------
// Depthwise K=11 'same' conv (channels-last, register ring over t).
// mode 0: Zh = dw; FINAL = dw * gate[b,0]
// mode 1: Y = dw
// mode 2: z = dw * Y; Zh = z; FINAL += z * gate[b,dIdx]
// grid (T/32, B), block 256 (= c); fully coalesced along c.
// ---------------------------------------------------------------------------
#define DW_TT 32
__global__ __launch_bounds__(256) void k_dw(
    const float* __restrict__ INP, const float* __restrict__ kw,
    const float* __restrict__ kb, const float* __restrict__ gate,
    _Float16* __restrict__ Zh, float* __restrict__ Yb,
    float* __restrict__ FINAL, int dIdx, int mode) {
  const int c = threadIdx.x;
  const int t0 = blockIdx.x * DW_TT;
  const int b = blockIdx.y;
  const size_t bb = (size_t)b * Tn;

  float w[Kk];
#pragma unroll
  for (int j = 0; j < Kk; ++j) w[j] = kw[c * Kk + j];
  const float bc = kb[c];
  const float g = (mode == 1) ? 0.f : gate[b * Dn + ((mode == 0) ? 0 : dIdx)];

  float ring[Kk];
#pragma unroll
  for (int j = 0; j < Kk; ++j) {
    int tt = t0 - (Kk / 2) + j;
    ring[j] = (tt >= 0 && tt < Tn) ? INP[(bb + tt) * Hn + c] : 0.0f;
  }
  for (int t = t0; t < t0 + DW_TT; ++t) {
    float acc = bc;
#pragma unroll
    for (int j = 0; j < Kk; ++j) acc += ring[j] * w[j];
    size_t idx = (bb + t) * Hn + c;
    if (mode == 0) {
      Zh[idx] = (_Float16)acc;
      FINAL[idx] = acc * g;
    } else if (mode == 1) {
      Yb[idx] = acc;
    } else {
      float zv = acc * Yb[idx];
      Zh[idx] = (_Float16)zv;
      FINAL[idx] += zv * g;
    }
#pragma unroll
    for (int j = 0; j < Kk - 1; ++j) ring[j] = ring[j + 1];
    int tn = t + 1 + (Kk / 2);
    ring[Kk - 1] = (tn < Tn) ? INP[(bb + tn) * Hn + c] : 0.0f;
  }
}

// ---------------------------------------------------------------------------
// LayerNorm over channels of (A + R) at each (b,t); channels-last so fully
// contiguous. grid B*T, block 256(=H). Safe for OUT aliasing A or R.
// ---------------------------------------------------------------------------
__global__ __launch_bounds__(256) void k_ln(
    const float* __restrict__ A, const float* __restrict__ R,
    const float* __restrict__ g, const float* __restrict__ beta,
    float* __restrict__ OUT, _Float16* __restrict__ OUTh) {
  __shared__ float sbuf[256];
  const int c = threadIdx.x;
  const size_t idx = (size_t)blockIdx.x * Hn + c;
  float v = A[idx] + R[idx];
  float mu = block_reduce_sum(v, sbuf) * (1.0f / Hn);
  float dv = v - mu;
  float var = block_reduce_sum(dv * dv, sbuf) * (1.0f / Hn);
  float o = dv * rsqrtf(var + EPSc) * g[c] + beta[c];
  OUT[idx] = o;
  OUTh[idx] = (_Float16)o;
}

// ---------------------------------------------------------------------------
// Mean pool over T, two-stage. POOLP[b][16][256] partials.
// ---------------------------------------------------------------------------
__global__ __launch_bounds__(256) void k_pool1(
    const float* __restrict__ X, float* __restrict__ POOLP) {
  const int c = threadIdx.x;
  const int ch = blockIdx.x;
  const int b = blockIdx.y;
  const size_t bb = (size_t)b * Tn;
  const int t0 = ch * FT_CH;
  float s = 0.f;
  for (int t = t0; t < t0 + FT_CH; ++t) s += X[(bb + t) * Hn + c];
  POOLP[((size_t)b * 16 + ch) * Hn + c] = s;
}

__global__ __launch_bounds__(256) void k_pool2(
    const float* __restrict__ POOLP, float* __restrict__ P) {
  const int c = threadIdx.x;
  const int b = blockIdx.x;
  float s = 0.f;
  for (int ch = 0; ch < 16; ++ch) s += POOLP[((size_t)b * 16 + ch) * Hn + c];
  P[(size_t)b * Hn + c] = s / (float)Tn;
}

// ---------------------------------------------------------------------------
// Classifier head: linear(H->H) + LN + GELU + linear(H->NCLS). grid B, blk 256
// ---------------------------------------------------------------------------
__global__ __launch_bounds__(256) void k_head(
    const float* __restrict__ P, const float* __restrict__ w1,
    const float* __restrict__ b1, const float* __restrict__ lng,
    const float* __restrict__ lnb, const float* __restrict__ w2,
    const float* __restrict__ b2, float* __restrict__ OUT) {
  __shared__ float sbuf[256];
  __shared__ float hs[256];
  const int b = blockIdx.x;
  const int m = threadIdx.x;
  const float* p = P + (size_t)b * Hn;
  float h1 = b1[m];
  for (int k = 0; k < Hn; ++k) h1 += p[k] * w1[(size_t)m * Hn + k];
  float mu = block_reduce_sum(h1, sbuf) * (1.0f / Hn);
  float dv = h1 - mu;
  float var = block_reduce_sum(dv * dv, sbuf) * (1.0f / Hn);
  float hn = dv * rsqrtf(var + EPSc) * lng[m] + lnb[m];
  hs[m] = gelu_f(hn);
  __syncthreads();
  if (m < NCLSn) {
    float o = b2[m];
    for (int k = 0; k < Hn; ++k) o += hs[k] * w2[(size_t)m * Hn + k];
    OUT[(size_t)b * NCLSn + m] = o;
  }
}

// ---------------------------------------------------------------------------
// Host launcher
// ---------------------------------------------------------------------------
extern "C" void kernel_launch(void* const* d_in, const int* in_sizes, int n_in,
                              void* d_out, int out_size, void* d_ws, size_t ws_size,
                              hipStream_t stream) {
  (void)in_sizes; (void)n_in; (void)out_size; (void)ws_size;
  const float* x         = (const float*)d_in[0];
  const float* in_w      = (const float*)d_in[1];
  const float* in_b      = (const float*)d_in[2];
  const float* gate_w    = (const float*)d_in[3];
  const float* gate_b    = (const float*)d_in[4];
  const float* ch_w      = (const float*)d_in[5];
  const float* ch_b      = (const float*)d_in[6];
  const float* tok_w     = (const float*)d_in[7];
  const float* tok_b     = (const float*)d_in[8];
  const float* pre_ch_w  = (const float*)d_in[9];
  const float* pre_ch_b  = (const float*)d_in[10];
  const float* pre_tok_w = (const float*)d_in[11];
  const float* pre_tok_b = (const float*)d_in[12];
  const float* ffn_w1    = (const float*)d_in[13];
  const float* ffn_b1    = (const float*)d_in[14];
  const float* ffn_w2    = (const float*)d_in[15];
  const float* ffn_b2    = (const float*)d_in[16];
  const float* ln1_g     = (const float*)d_in[17];
  const float* ln1_b     = (const float*)d_in[18];
  const float* ln2_g     = (const float*)d_in[19];
  const float* ln2_b     = (const float*)d_in[20];
  const float* cls_w1    = (const float*)d_in[21];
  const float* cls_b1    = (const float*)d_in[22];
  const float* cls_ln_g  = (const float*)d_in[23];
  const float* cls_ln_b  = (const float*)d_in[24];
  const float* cls_w2    = (const float*)d_in[25];
  const float* cls_b2    = (const float*)d_in[26];

  // Workspace. S = B*T*H = 16,777,216 elements.
  const size_t S = (size_t)Bsz * Tn * Hn;
  float* X     = (float*)d_ws;
  float* FINAL = X + S;
  float* A     = FINAL + S;
  float* Y     = A + S;
  _Float16* Xh   = (_Float16*)(Y + S);
  _Float16* Zh   = Xh + S;
  _Float16* Wch  = Zh + S;
  _Float16* Wpre = Wch + (size_t)Ln * Dn * Hn * Hn;
  _Float16* Wf1  = Wpre + (size_t)Ln * (Dn - 1) * Hn * Hn;
  _Float16* Wf2  = Wf1 + (size_t)Ln * 2 * Hn * Hn;
  float* FEAT    = (float*)(Wf2 + (size_t)Ln * 2 * Hn * Hn);
  float* GATE    = FEAT + (size_t)Bsz * 3 * Hn;
  float* POOLED  = GATE + (size_t)Bsz * Dn;
  float* FEATP   = POOLED + (size_t)Bsz * Hn;
  float* POOLP   = FEATP + (size_t)Bsz * 16 * 3 * Hn;
  _Float16* HFFNh = (_Float16*)Y;  // 2S halves over the Y region

  const dim3 blk(256);
  const dim3 dwGrid(Tn / DW_TT, Bsz);
  const dim3 lnGrid(Bsz * Tn);
  const dim3 gemmHH(Tn / 128, Hn / 128, Bsz);

  // One-time f16 weight conversion.
  {
    int n1 = Ln * Dn * Hn * Hn;
    int n2 = Ln * (Dn - 1) * Hn * Hn;
    int n3 = Ln * 2 * Hn * Hn;
    k_cvt<<<dim3(n1 / 256), blk, 0, stream>>>(ch_w, Wch, n1);
    k_cvt<<<dim3(n2 / 256), blk, 0, stream>>>(pre_ch_w, Wpre, n2);
    k_cvt<<<dim3(n3 / 256), blk, 0, stream>>>(ffn_w1, Wf1, n3);
    k_cvt<<<dim3(n3 / 256), blk, 0, stream>>>(ffn_w2, Wf2, n3);
  }

  // Stem
  k_pw_in<<<dim3(Tn, Bsz), blk, 0, stream>>>(x, in_w, in_b, X, Xh);

  for (int i = 0; i < Ln; ++i) {
    // Gate
    k_feat1<<<dim3(16, Bsz), blk, 0, stream>>>(X, FEATP);
    k_feat2<<<dim3(Bsz), blk, 0, stream>>>(FEATP, FEAT);
    k_logits<<<dim3(1), dim3(64), 0, stream>>>(
        FEAT, gate_w + (size_t)i * Dn * 3 * Hn, gate_b + i * Dn, GATE);

    for (int d = 0; d < Dn; ++d) {
      k_pw_gemm<<<gemmHH, blk, 0, stream>>>(
          Xh, Wch + (size_t)(i * Dn + d) * Hn * Hn, ch_b + (i * Dn + d) * Hn,
          A, nullptr, Hn, Hn, 0, 0);
      if (d == 0) {
        k_dw<<<dwGrid, blk, 0, stream>>>(
            A, tok_w + (size_t)(i * Dn) * Hn * Kk, tok_b + (i * Dn) * Hn,
            GATE, Zh, Y, FINAL, 0, 0);
      } else {
        k_dw<<<dwGrid, blk, 0, stream>>>(
            A, tok_w + (size_t)(i * Dn + d) * Hn * Kk,
            tok_b + (i * Dn + d) * Hn, GATE, Zh, Y, FINAL, d, 1);
        k_pw_gemm<<<gemmHH, blk, 0, stream>>>(
            Zh, Wpre + (size_t)(i * (Dn - 1) + d - 1) * Hn * Hn,
            pre_ch_b + (i * (Dn - 1) + d - 1) * Hn, A, nullptr, Hn, Hn, 0, 0);
        k_dw<<<dwGrid, blk, 0, stream>>>(
            A, pre_tok_w + (size_t)(i * (Dn - 1) + d - 1) * Hn * Kk,
            pre_tok_b + (i * (Dn - 1) + d - 1) * Hn, GATE, Zh, Y, FINAL, d, 2);
      }
    }
    // LN1(final + res) -> X, Xh
    k_ln<<<lnGrid, blk, 0, stream>>>(FINAL, X, ln1_g + i * Hn, ln1_b + i * Hn,
                                     X, Xh);

    // FFN: Xh -> HFFNh (gelu, f16) -> A; LN2(A + X) -> X, Xh
    k_pw_gemm<<<dim3(Tn / 128, (2 * Hn) / 128, Bsz), blk, 0, stream>>>(
        Xh, Wf1 + (size_t)i * 2 * Hn * Hn, ffn_b1 + i * 2 * Hn,
        nullptr, HFFNh, 2 * Hn, Hn, 1, 1);
    k_pw_gemm<<<gemmHH, blk, 0, stream>>>(
        HFFNh, Wf2 + (size_t)i * Hn * 2 * Hn, ffn_b2 + i * Hn, A, nullptr,
        Hn, 2 * Hn, 0, 0);
    k_ln<<<lnGrid, blk, 0, stream>>>(A, X, ln2_g + i * Hn, ln2_b + i * Hn,
                                     X, Xh);
  }

  // Head
  k_pool1<<<dim3(16, Bsz), blk, 0, stream>>>(X, POOLP);
  k_pool2<<<dim3(Bsz), blk, 0, stream>>>(POOLP, POOLED);
  k_head<<<dim3(Bsz), blk, 0, stream>>>(
      POOLED, cls_w1, cls_b1, cls_ln_g, cls_ln_b, cls_w2, cls_b2,
      (float*)d_out);
}
